// EGCL_60954175865324
// MI455X (gfx1250) — matrix operand order repaired and checked by reading, not compiled, not run on hardware
//
#include <hip/hip_runtime.h>
#include <hip/hip_bf16.h>

// ---------------------------------------------------------------------------
// EGCL (EGNN layer) fused for MI455X / gfx1250 (wave32, WMMA bf16 -> f32 acc)
// ---------------------------------------------------------------------------

typedef __bf16 bf16_t;
typedef __attribute__((ext_vector_type(16))) __bf16 v16bf;
typedef __attribute__((ext_vector_type(8)))  float  v8f;

#define ND 128      // node feature dim
#define MD 128      // message / hidden dim
#define K1 256      // 2*ND (radial column handled as rank-1 epilogue)
#define BE 64       // edges per block tile
#define BN 64       // nodes per block tile

union Frag16 { v16bf v; float4 f4[2]; };

__device__ __forceinline__ float silu_f(float x) {
    return x / (1.0f + __expf(-x));
}

// A-fragment (16x32 bf16, M x K) from LDS, row-major with stride == K.
// ISA layout: lane L -> M = L&15 ; kb = (L>=16)?8:0 ;
//   elements 0..7  = K kb..kb+7   (VGPR0..3)
//   elements 8..15 = K kb+16..kb+23 (VGPR4..7)
__device__ __forceinline__ v16bf load_a_frag_lds(const bf16_t* base, int row0,
                                                 int stride, int ks) {
    int lane = threadIdx.x & 31;
    int m    = lane & 15;
    int kb   = (lane & 16) ? 8 : 0;
    const bf16_t* p = base + (row0 + m) * stride + ks + kb;
    Frag16 f;
    f.f4[0] = *reinterpret_cast<const float4*>(p);        // K kb..kb+7
    f.f4[1] = *reinterpret_cast<const float4*>(p + 16);   // K kb+16..kb+23
    return f.v;
}

// B-fragment (32x16 bf16, K x N) from global row-major weight W[outN][K]:
// lane L -> N = L&15 ; kb = (L>=16)?8:0 ; elements 0..7 = K kb..kb+7, 8..15 = K kb+16..kb+23.
__device__ __forceinline__ v16bf load_b_frag_g(const bf16_t* W, int K,
                                               int n0, int ks) {
    int lane = threadIdx.x & 31;
    int n    = lane & 15;
    int kb   = (lane & 16) ? 8 : 0;
    const bf16_t* p = W + (size_t)(n0 + n) * K + ks + kb;
    Frag16 f;
    f.f4[0] = *reinterpret_cast<const float4*>(p);        // K kb..kb+7
    f.f4[1] = *reinterpret_cast<const float4*>(p + 16);   // K kb+16..kb+23
    return f.v;
}

// One 16-row x 64-col tile: A (LDS, stride==K) x W^T (global bf16) -> 4 accs.
template <int K>
__device__ __forceinline__ void gemm_tile(const bf16_t* A, int mBase,
                                          const bf16_t* W, int nBase,
                                          v8f acc[4]) {
#pragma unroll
    for (int ks = 0; ks < K; ks += 32) {
        v16bf a = load_a_frag_lds(A, mBase, K, ks);
#pragma unroll
        for (int j = 0; j < 4; ++j) {
            v16bf b = load_b_frag_g(W, K, nBase + 16 * j, ks);
            acc[j] = __builtin_amdgcn_wmma_f32_16x16x32_bf16(
                false, a, false, b, (short)0, acc[j], false, false);
        }
    }
}

// ---------------------------------------------------------------------------
// Kernel 0a: zero accumulators (ws is not re-poisoned between replays)
// ---------------------------------------------------------------------------
__global__ void egcl_zero_kernel(float* m_agg, float* coord_agg, float* counts,
                                 int N) {
    int i = blockIdx.x * 256 + threadIdx.x;
    if (i < N * ND) m_agg[i] = 0.0f;
    if (i < N * 3)  coord_agg[i] = 0.0f;
    if (i < N)      counts[i] = 0.0f;
}

// ---------------------------------------------------------------------------
// Kernel 0b: fp32 weights -> bf16 (L2-resident, reused by all blocks)
// ---------------------------------------------------------------------------
__global__ void egcl_convert_kernel(const float* We1, const float* We2,
                                    const float* Wc1, const float* Wn1,
                                    const float* Wn2,
                                    bf16_t* gWe1, float* wR1, bf16_t* gWe2,
                                    bf16_t* gWc1, bf16_t* gWn1, bf16_t* gWn2) {
    int i = blockIdx.x * 256 + threadIdx.x;
    if (i < MD * K1) {               // We1 main block [128][256] (drop col 256)
        int r = i >> 8, c = i & 255;
        gWe1[i] = (bf16_t)We1[r * 257 + c];
    }
    if (i < MD) wR1[i] = We1[i * 257 + 256];   // radial column, kept fp32
    if (i < MD * MD) gWe2[i] = (bf16_t)We2[i];
    if (i < MD * MD) gWc1[i] = (bf16_t)Wc1[i];
    if (i < MD * K1) gWn1[i] = (bf16_t)Wn1[i]; // Wn1 [128][256]
    if (i < MD * MD) gWn2[i] = (bf16_t)Wn2[i];
}

// ---------------------------------------------------------------------------
// Kernel 1: fused edge pipeline
//   gather -> GEMM1(+radial,SiLU) -> GEMM2(SiLU, scatter m_agg)
//          -> GEMM3(SiLU) -> scale dot -> coord_agg/counts atomics
// ---------------------------------------------------------------------------
__global__ void __launch_bounds__(256)
egcl_edge_kernel(const float* __restrict__ coord,
                 const float* __restrict__ node_feat,
                 const int*   __restrict__ eidx,
                 const float* __restrict__ be1, const float* __restrict__ be2,
                 const float* __restrict__ bc1, const float* __restrict__ Wc2,
                 const bf16_t* __restrict__ gWe1, const float* __restrict__ wR1,
                 const bf16_t* __restrict__ gWe2, const bf16_t* __restrict__ gWc1,
                 float* m_agg, float* coord_agg, float* counts, int E) {
    extern __shared__ char smem[];
    bf16_t* sFeat = (bf16_t*)(smem);            // 64 x 256 bf16  (32768 B)
    bf16_t* sM1   = (bf16_t*)(smem + 32768);    // 64 x 128 bf16  (16384 B)
    bf16_t* sM2   = (bf16_t*)(smem + 49152);    // 64 x 128 bf16
    bf16_t* sT1   = (bf16_t*)(smem + 65536);    // 64 x 128 bf16
    int*    sRow  = (int*)   (smem + 81920);    // 64
    int*    sCol  = (int*)   (smem + 82176);    // 64
    float*  sRad  = (float*) (smem + 82432);    // 64
    float*  sDiff = (float*) (smem + 82688);    // 64 x 3

    const int tid = threadIdx.x;
    const int e0  = blockIdx.x * BE;

    // --- edge metadata: indices, coord_diff, radial -------------------------
    if (tid < BE) {
        int e = e0 + tid;
        int r = 0, c = 0;
        float d0 = 0.f, d1 = 0.f, d2 = 0.f, rad = 0.f;
        bool valid = (e < E);
        if (valid) {
            r  = eidx[e];
            c  = eidx[E + e];
            d0 = coord[r * 3 + 0] - coord[c * 3 + 0];
            d1 = coord[r * 3 + 1] - coord[c * 3 + 1];
            d2 = coord[r * 3 + 2] - coord[c * 3 + 2];
            rad = d0 * d0 + d1 * d1 + d2 * d2;
        }
        sRow[tid] = valid ? r : -1;
        sCol[tid] = c;
        sRad[tid] = rad;
        sDiff[tid * 3 + 0] = d0;
        sDiff[tid * 3 + 1] = d1;
        sDiff[tid * 3 + 2] = d2;
    }
    __syncthreads();

    // --- gather [h_row | h_col] into bf16 tile (float4 vectorized) ----------
    for (int idx = tid; idx < BE * (K1 / 4); idx += 256) {
        int e  = idx >> 6;               // 64 float4 per edge row
        int k4 = (idx & 63) * 4;
        int rI = sRow[e]; if (rI < 0) rI = 0;
        int src = (k4 < ND) ? rI : sCol[e];
        int kk  = (k4 < ND) ? k4 : (k4 - ND);
        float4 v = *reinterpret_cast<const float4*>(node_feat + (size_t)src * ND + kk);
        bf16_t* dst = sFeat + e * K1 + k4;
        dst[0] = (bf16_t)v.x; dst[1] = (bf16_t)v.y;
        dst[2] = (bf16_t)v.z; dst[3] = (bf16_t)v.w;
    }
    __syncthreads();

    const int wave  = tid >> 5;
    const int lane  = tid & 31;
    const int mT    = (wave & 3) * 16;          // row tile base (edge in tile)
    const int nHalf = (wave >> 2) * 64;         // column half base
    const int rOff  = (lane & 16) ? 8 : 0;      // D-layout M offset
    const int nl    = lane & 15;                // D-layout N

    // --- GEMM1: edge_feat @ We1^T (+ bias + radial rank-1), SiLU ------------
    {
        v8f acc[4] = {};
        gemm_tile<K1>(sFeat, mT, gWe1, nHalf, acc);
#pragma unroll
        for (int j = 0; j < 4; ++j) {
            int col = nHalf + j * 16 + nl;
            float bcol = be1[col];
            float wr   = wR1[col];
#pragma unroll
            for (int r = 0; r < 8; ++r) {
                int eL  = mT + r + rOff;
                float v = acc[j][r] + bcol + sRad[eL] * wr;
                sM1[eL * MD + col] = (bf16_t)silu_f(v);
            }
        }
    }
    __syncthreads();

    // --- GEMM2: h1 @ We2^T, SiLU -> m ; scatter-add into m_agg --------------
    {
        v8f acc[4] = {};
        gemm_tile<MD>(sM1, mT, gWe2, nHalf, acc);
#pragma unroll
        for (int j = 0; j < 4; ++j) {
            int col = nHalf + j * 16 + nl;
            float bcol = be2[col];
#pragma unroll
            for (int r = 0; r < 8; ++r) {
                int eL  = mT + r + rOff;
                float v = silu_f(acc[j][r] + bcol);
                sM2[eL * MD + col] = (bf16_t)v;
                int rI = sRow[eL];
                if (rI >= 0) atomicAdd(&m_agg[(size_t)rI * MD + col], v);
            }
        }
    }
    __syncthreads();

    // --- GEMM3: m @ Wc1^T, SiLU -> t1 ---------------------------------------
    {
        v8f acc[4] = {};
        gemm_tile<MD>(sM2, mT, gWc1, nHalf, acc);
#pragma unroll
        for (int j = 0; j < 4; ++j) {
            int col = nHalf + j * 16 + nl;
            float bcol = bc1[col];
#pragma unroll
            for (int r = 0; r < 8; ++r) {
                int eL = mT + r + rOff;
                sT1[eL * MD + col] = (bf16_t)silu_f(acc[j][r] + bcol);
            }
        }
    }
    __syncthreads();

    // --- scale = t1 . Wc2 ; trans = coord_diff * scale ; coord atomics ------
    if (tid < BE) {
        int rI = sRow[tid];
        if (rI >= 0) {
            float dot = 0.0f;
#pragma unroll 8
            for (int k = 0; k < MD; ++k)
                dot += (float)sT1[tid * MD + k] * Wc2[k];
            atomicAdd(&coord_agg[rI * 3 + 0], sDiff[tid * 3 + 0] * dot);
            atomicAdd(&coord_agg[rI * 3 + 1], sDiff[tid * 3 + 1] * dot);
            atomicAdd(&coord_agg[rI * 3 + 2], sDiff[tid * 3 + 2] * dot);
            atomicAdd(&counts[rI], 1.0f);
        }
    }
}

// ---------------------------------------------------------------------------
// Kernel 2: fused node update + coord output
//   h = [node_feat | m_agg] -> GEMM(Wn1)+SiLU -> GEMM(Wn2)+bias + residual
//   coord_out = coord + coord_agg / max(counts, 1)
// ---------------------------------------------------------------------------
__global__ void __launch_bounds__(256)
egcl_node_kernel(const float* __restrict__ coord,
                 const float* __restrict__ node_feat,
                 const float* __restrict__ bn1, const float* __restrict__ bn2,
                 const bf16_t* __restrict__ gWn1, const bf16_t* __restrict__ gWn2,
                 const float* __restrict__ m_agg,
                 const float* __restrict__ coord_agg,
                 const float* __restrict__ counts,
                 float* outCoord, float* outNode, int N) {
    extern __shared__ char smem[];
    bf16_t* sH = (bf16_t*)(smem);           // 64 x 256 bf16 (32768 B)
    bf16_t* sU = (bf16_t*)(smem + 32768);   // 64 x 128 bf16 (16384 B)

    const int tid = threadIdx.x;
    const int n0  = blockIdx.x * BN;

    // --- build h tile -------------------------------------------------------
    for (int idx = tid; idx < BN * (K1 / 4); idx += 256) {
        int nLoc = idx >> 6;
        int k4   = (idx & 63) * 4;
        int g = n0 + nLoc; if (g >= N) g = N - 1;
        const float* src = (k4 < ND) ? (node_feat + (size_t)g * ND + k4)
                                     : (m_agg + (size_t)g * MD + (k4 - ND));
        float4 v = *reinterpret_cast<const float4*>(src);
        bf16_t* dst = sH + nLoc * K1 + k4;
        dst[0] = (bf16_t)v.x; dst[1] = (bf16_t)v.y;
        dst[2] = (bf16_t)v.z; dst[3] = (bf16_t)v.w;
    }
    __syncthreads();

    const int wave  = tid >> 5;
    const int lane  = tid & 31;
    const int mT    = (wave & 3) * 16;
    const int nHalf = (wave >> 2) * 64;
    const int rOff  = (lane & 16) ? 8 : 0;
    const int nl    = lane & 15;

    // --- GEMM A: h @ Wn1^T + bn1, SiLU --------------------------------------
    {
        v8f acc[4] = {};
        gemm_tile<K1>(sH, mT, gWn1, nHalf, acc);
#pragma unroll
        for (int j = 0; j < 4; ++j) {
            int col = nHalf + j * 16 + nl;
            float bcol = bn1[col];
#pragma unroll
            for (int r = 0; r < 8; ++r) {
                int nLoc = mT + r + rOff;
                sU[nLoc * MD + col] = (bf16_t)silu_f(acc[j][r] + bcol);
            }
        }
    }
    __syncthreads();

    // --- GEMM B: u @ Wn2^T + bn2 + residual ---------------------------------
    {
        v8f acc[4] = {};
        gemm_tile<MD>(sU, mT, gWn2, nHalf, acc);
#pragma unroll
        for (int j = 0; j < 4; ++j) {
            int col = nHalf + j * 16 + nl;
            float bcol = bn2[col];
#pragma unroll
            for (int r = 0; r < 8; ++r) {
                int g = n0 + mT + r + rOff;
                if (g < N) {
                    float res = node_feat[(size_t)g * ND + col];
                    outNode[(size_t)g * ND + col] = res + acc[j][r] + bcol;
                }
            }
        }
    }

    // --- coord output -------------------------------------------------------
    if (tid < BN) {
        int g = n0 + tid;
        if (g < N) {
            float cnt = counts[g];
            if (cnt < 1.0f) cnt = 1.0f;
            float inv = 1.0f / cnt;
#pragma unroll
            for (int c = 0; c < 3; ++c)
                outCoord[g * 3 + c] = coord[g * 3 + c] + coord_agg[g * 3 + c] * inv;
        }
    }
}

// ---------------------------------------------------------------------------
// Host launcher
// ---------------------------------------------------------------------------
extern "C" void kernel_launch(void* const* d_in, const int* in_sizes, int n_in,
                              void* d_out, int out_size, void* d_ws, size_t ws_size,
                              hipStream_t stream) {
    const float* coord     = (const float*)d_in[0];
    const float* node_feat = (const float*)d_in[1];
    const int*   eidx      = (const int*)  d_in[2];
    const float* We1 = (const float*)d_in[3];
    const float* be1 = (const float*)d_in[4];
    const float* We2 = (const float*)d_in[5];
    const float* be2 = (const float*)d_in[6];
    const float* Wn1 = (const float*)d_in[7];
    const float* bn1 = (const float*)d_in[8];
    const float* Wn2 = (const float*)d_in[9];
    const float* bn2 = (const float*)d_in[10];
    const float* Wc1 = (const float*)d_in[11];
    const float* bc1 = (const float*)d_in[12];
    const float* Wc2 = (const float*)d_in[13];

    const int N = in_sizes[0] / 3;
    const int E = in_sizes[2] / 2;

    // workspace carve-up
    char* ws = (char*)d_ws;
    size_t off = 0;
    auto take = [&](size_t bytes) -> char* {
        char* p = ws + off;
        off += (bytes + 255) & ~(size_t)255;
        return p;
    };
    float*  m_agg     = (float*) take((size_t)N * MD * 4);
    float*  coord_agg = (float*) take((size_t)N * 3 * 4);
    float*  counts    = (float*) take((size_t)N * 4);
    bf16_t* gWe1      = (bf16_t*)take((size_t)MD * K1 * 2);
    float*  wR1       = (float*) take((size_t)MD * 4);
    bf16_t* gWe2      = (bf16_t*)take((size_t)MD * MD * 2);
    bf16_t* gWc1      = (bf16_t*)take((size_t)MD * MD * 2);
    bf16_t* gWn1      = (bf16_t*)take((size_t)MD * K1 * 2);
    bf16_t* gWn2      = (bf16_t*)take((size_t)MD * MD * 2);

    float* outCoord = (float*)d_out;
    float* outNode  = (float*)d_out + (size_t)N * 3;

    egcl_zero_kernel<<<(N * ND + 255) / 256, 256, 0, stream>>>(m_agg, coord_agg,
                                                               counts, N);
    egcl_convert_kernel<<<(MD * K1 + 255) / 256, 256, 0, stream>>>(
        We1, We2, Wc1, Wn1, Wn2, gWe1, wR1, gWe2, gWc1, gWn1, gWn2);

    const size_t edgeLds = 83456;  // tiles + metadata, matches kernel offsets
    egcl_edge_kernel<<<(E + BE - 1) / BE, 256, edgeLds, stream>>>(
        coord, node_feat, eidx, be1, be2, bc1, Wc2, gWe1, wR1, gWe2, gWc1,
        m_agg, coord_agg, counts, E);

    const size_t nodeLds = 32768 + 16384;
    egcl_node_kernel<<<(N + BN - 1) / BN, 256, nodeLds, stream>>>(
        coord, node_feat, bn1, bn2, gWn1, gWn2, m_agg, coord_agg, counts,
        outCoord, outNode, N);
}
